// NaiveFourierKANLayer_30159260352766
// MI455X (gfx1250) — matrix operand order repaired and checked by reading, not compiled, not run on hardware
//
#include <hip/hip_runtime.h>

typedef __attribute__((ext_vector_type(16))) _Float16 v16h;
typedef __attribute__((ext_vector_type(8)))  float    v8f;

#define NROWS   32768
#define IN_DIM  64
#define OUT_DIM 256
#define GRIDSZ  32
#define KFEAT   4096          // 2 * IN_DIM * GRIDSZ
#define KTILES  128           // KFEAT / 32
#define NTILES  16            // OUT_DIM / 16
#define WAVES   4

// ---------------------------------------------------------------------------
// Kernel 1: pack fouriercoeffs (2,256,64,32) f32 -> f16 WMMA B-fragments in ws.
// Fragment storage: frag (kt,nt) occupies 32 lanes x 16 halfs contiguous:
//   ws[ ((kt*16 + nt)*32 + lane)*16 + j ]
// B 32x16 layout (this code's contiguous k order, same as A): N = lane&15,
//   K = j + (lane>=16 ? 16 : 0)  within the 32-wide K tile.
// Feature index f = kt*32 + K  maps to (trig = f>>11, i = (f>>5)&63, g = f&31).
// ---------------------------------------------------------------------------
__global__ __launch_bounds__(256) void fkan_pack_b(
    const float* __restrict__ coeffs, _Float16* __restrict__ bw) {
  int tid = blockIdx.x * 256 + threadIdx.x;      // 0 .. KFEAT*OUT_DIM-1
  if (tid >= KFEAT * OUT_DIM) return;
  int j    =  tid        & 15;
  int lane = (tid >> 4)  & 31;
  int nt   = (tid >> 9)  & 15;
  int kt   =  tid >> 13;                         // 0..127
  int kl   = j + ((lane >= 16) ? 16 : 0);        // K within tile
  int f    = kt * 32 + kl;                       // global feature index
  int o    = nt * 16 + (lane & 15);              // output column
  int trig =  f >> 11;                           // 0=cos bank, 1=sin bank
  int i    = (f >> 5) & (IN_DIM - 1);
  int g    =  f & (GRIDSZ - 1);
  bw[tid] = (_Float16)coeffs[((trig * OUT_DIM + o) * IN_DIM + i) * GRIDSZ + g];
}

// ---------------------------------------------------------------------------
// A-fragment generation: 16x32 f16, contiguous k order (same as B pack).
//   lanes 0-15 : j=0..15 -> K=j     (K 0-15)
//   lanes 16-31: j=0..15 -> K=j+16  (K 16-31)
// K within tile == grid index g; element = trig(xv * (g+1)).
// ---------------------------------------------------------------------------
template <bool USE_SIN>
__device__ __forceinline__ v16h make_a_frag(float xv, int hi) {
  v16h a;
#pragma unroll
  for (int j = 0; j < 16; ++j) {
    int g = j + hi * 16;
    float ang = xv * (float)(g + 1);
    float t = USE_SIN ? __sinf(ang) : __cosf(ang);
    a[j] = (_Float16)t;
  }
  return a;
}

// ---------------------------------------------------------------------------
// Kernel 2: GEMM. Each wave owns a 16-row strip and all 256 output columns:
// 16 accumulators (v8f each), 128 K-tiles, 16 WMMAs per K-tile.
// ---------------------------------------------------------------------------
__global__ __launch_bounds__(WAVES * 32) void fkan_gemm(
    const float* __restrict__ x, const _Float16* __restrict__ bw,
    const float* __restrict__ bias, float* __restrict__ out) {
  const int wave  = threadIdx.x >> 5;
  const int lane  = threadIdx.x & 31;
  const int strip = blockIdx.x * WAVES + wave;   // 0..2047
  const int hi    = lane >> 4;                   // lane half
  const int row   = strip * 16 + (lane & 15);

  v8f acc[NTILES];
#pragma unroll
  for (int nt = 0; nt < NTILES; ++nt)
    acc[nt] = (v8f){0.f, 0.f, 0.f, 0.f, 0.f, 0.f, 0.f, 0.f};

  const float* xr    = x + row * IN_DIM;
  const v16h*  bfrag = (const v16h*)bw;          // fragment-major, see pack

  // Phase 0: cos bank (kt = 0..63), Phase 1: sin bank (kt = 64..127).
#pragma unroll 1
  for (int kt = 0; kt < 64; ++kt) {
    float xv = xr[kt];
    v16h a = make_a_frag<false>(xv, hi);
    const v16h* bt = bfrag + (size_t)kt * NTILES * 32 + lane;
#pragma unroll
    for (int nt = 0; nt < NTILES; ++nt) {
      v16h b = bt[nt * 32];
      acc[nt] = __builtin_amdgcn_wmma_f32_16x16x32_f16(
          false, a, false, b, (short)0, acc[nt], false, false);
    }
  }
#pragma unroll 1
  for (int kt = 64; kt < 128; ++kt) {
    float xv = xr[kt - 64];
    v16h a = make_a_frag<true>(xv, hi);
    const v16h* bt = bfrag + (size_t)kt * NTILES * 32 + lane;
#pragma unroll
    for (int nt = 0; nt < NTILES; ++nt) {
      v16h b = bt[nt * 32];
      acc[nt] = __builtin_amdgcn_wmma_f32_16x16x32_f16(
          false, a, false, b, (short)0, acc[nt], false, false);
    }
  }

  // C/D layout (ISA 7.12.2): VGPR r, lanes 0-15 -> M=r, lanes 16-31 -> M=r+8,
  // N = lane&15. Add bias and store f32.
#pragma unroll
  for (int nt = 0; nt < NTILES; ++nt) {
    int col = nt * 16 + (lane & 15);
    float bv = bias[col];
    float* op = out + (size_t)(strip * 16 + hi * 8) * OUT_DIM + col;
#pragma unroll
    for (int r = 0; r < 8; ++r)
      op[(size_t)r * OUT_DIM] = acc[nt][r] + bv;
  }
}

// ---------------------------------------------------------------------------
extern "C" void kernel_launch(void* const* d_in, const int* in_sizes, int n_in,
                              void* d_out, int out_size, void* d_ws, size_t ws_size,
                              hipStream_t stream) {
  const float* x      = (const float*)d_in[0];   // (32768, 64)
  const float* coeffs = (const float*)d_in[1];   // (2, 256, 64, 32)
  const float* bias   = (const float*)d_in[2];   // (1, 256)
  float*       out    = (float*)d_out;           // (32768, 256)
  _Float16*    bw     = (_Float16*)d_ws;         // 4096*256 f16 = 2 MB

  // 1) Re-pack weights into WMMA fragment order (runs every call; ws is
  //    fully overwritten, so no cross-call state).
  {
    int total  = KFEAT * OUT_DIM;                // 1,048,576
    int blocks = (total + 255) / 256;            // 4096
    fkan_pack_b<<<blocks, 256, 0, stream>>>(coeffs, bw);
  }

  // 2) WMMA GEMM: 2048 strips of 16 rows, 4 waves per block.
  {
    int blocks = (NROWS / 16) / WAVES;           // 512
    fkan_gemm<<<blocks, WAVES * 32, 0, stream>>>(x, bw, bias, out);
  }
}
